// ShiftLSTMLayer_66314295050363
// MI455X (gfx1250) — matrix operand order repaired and checked by reading, not compiled, not run on hardware
//
#include <hip/hip_runtime.h>
#include <stdint.h>

#define Bdim  256
#define Tdim  512
#define Idim  128
#define Hdim  256
#define Sdim  4
#define NT    16          // batch tile per workgroup
#define Kdim  384         // I + H
#define GATES 1024        // 4*H

// LDS row strides (elements), padded so 16 rows hit distinct bank groups
#define XSTR  136         // ushort (bf16): 128 data + 8 pad  (68 dwords, mod 64 = 4)
#define HSTR  264         // ushort (bf16): 256 data + 8 pad  (132 dwords, mod 64 = 4)
#define GSTR  1028        // float: 1024 data + 4 pad        (1028 dwords, mod 64 = 4)

// LDS byte offsets (all 16B aligned)
#define G_OFF    0
#define H_OFF    (G_OFF + NT * GSTR * 4)        // 65792
#define X0_OFF   (H_OFF + NT * HSTR * 2)        // 74240
#define X1_OFF   (X0_OFF + NT * XSTR * 2)       // 78592
#define BIAS_OFF (X1_OFF + NT * XSTR * 2)       // 82944
#define LEN_OFF  (BIAS_OFF + Sdim * GATES * 4)  // 99328
#define SEG_OFF  (LEN_OFF + NT * 4)             // 99392
#define SMEM_BYTES (SEG_OFF + 2 * NT)           // 99424

typedef unsigned short ushort_t;
typedef __attribute__((ext_vector_type(16))) __bf16 v16bf;
typedef __attribute__((ext_vector_type(8)))  __bf16 bf16x8;
typedef __attribute__((ext_vector_type(8)))  float  v8f;
typedef __attribute__((ext_vector_type(4)))  unsigned int v4u;
typedef __attribute__((ext_vector_type(8)))  int v8i;
typedef __attribute__((ext_vector_type(4)))  int v4i;

__device__ __forceinline__ unsigned short f2bf(float f) {
    union { float f; unsigned int u; } v; v.f = f;
    unsigned int u = v.u;
    unsigned int r = u + 0x7fffu + ((u >> 16) & 1u);    // round-to-nearest-even
    if ((u & 0x7f800000u) == 0x7f800000u) r = u;        // inf/nan passthrough
    return (unsigned short)(r >> 16);
}

__device__ __forceinline__ float sigm(float x) {
    x = fminf(fmaxf(x, -30.f), 30.f);
    return 1.f / (1.f + __expf(-x));
}
__device__ __forceinline__ float tanh_f(float x) {
    x = fminf(fmaxf(x, -15.f), 15.f);
    float e = __expf(2.f * x);
    return (e - 1.f) / (e + 1.f);
}

// Load 16 bf16 from two 16-byte chunks into a v16bf (elements 0..7 <- p0, 8..15 <- p1)
__device__ __forceinline__ v16bf ld16(const ushort_t* p0, const ushort_t* p1) {
    bf16x8 lo = *(const bf16x8*)p0;
    bf16x8 hi = *(const bf16x8*)p1;
    return __builtin_shufflevector(lo, hi, 0,1,2,3,4,5,6,7,8,9,10,11,12,13,14,15);
}

// ---- TDM: async load one x-tile (16 rows x 128 bf16, row stride T*I elems) into LDS,
//      with LDS padding of 4 dwords after every 64 data dwords (matches XSTR=136).
__device__ __forceinline__ void tdm_load_x(const ushort_t* gptr, unsigned int lds_off) {
#if __has_builtin(__builtin_amdgcn_tensor_load_to_lds)
    unsigned long long ga = (unsigned long long)(uintptr_t)gptr;
    v4u g0;
    g0[0] = 1u;                                           // count=1, user mode
    g0[1] = lds_off;                                      // lds_addr (bytes)
    g0[2] = (unsigned int)(ga & 0xffffffffu);             // global_addr[31:0]
    g0[3] = (unsigned int)((ga >> 32) & 0x01ffffffu)      // global_addr[56:32]
          | 0x80000000u;                                  // type=2 ("image")
    v8i g1;
    g1[0] = (int)((1u << 16)        // data_size = 2 bytes
                | (1u << 20)        // pad_enable
                | (5u << 22)        // pad_interval: 64 dwords
                | (3u << 25));      // pad_amount: 4 dwords
    g1[1] = (int)((unsigned)(Idim & 0xffff) << 16);       // tensor_dim0[15:0] = 128
    g1[2] = (int)((unsigned)(NT & 0xffff) << 16);         // dim0 hi=0 | tensor_dim1[15:0]=16
    g1[3] = (int)((unsigned)(Idim & 0xffff) << 16);       // dim1 hi=0 | tile_dim0=128
    g1[4] = (int)(NT & 0xffff);                           // tile_dim1=16 | tile_dim2=0
    g1[5] = (int)(Tdim * Idim);                           // tensor_dim0_stride[31:0] = 65536
    g1[6] = 0;                                            // stride hi | dim1_stride lo
    g1[7] = 0;
    v4i g2 = {0, 0, 0, 0};                                // 2D tile: groups 2/3 unused
    v4i g3 = {0, 0, 0, 0};
    v8i g4 = {0, 0, 0, 0, 0, 0, 0, 0};                    // extra group (6-arg toolchain)
    __builtin_amdgcn_tensor_load_to_lds(g0, g1, g2, g3, g4, 0);
#else
    (void)gptr; (void)lds_off;
#endif
}

__device__ __forceinline__ void wait_tensor0() {
#if __has_builtin(__builtin_amdgcn_s_wait_tensorcnt)
    __builtin_amdgcn_s_wait_tensorcnt(0);
#else
    asm volatile("s_wait_tensorcnt 0x0" ::: "memory");
#endif
}

// ---------------------------------------------------------------------------
// Prep kernels
// ---------------------------------------------------------------------------
__global__ void prep_len(const unsigned char* __restrict__ mask, int* __restrict__ lenw) {
    int b = blockIdx.x * blockDim.x + threadIdx.x;
    if (b >= Bdim) return;
    const unsigned int* m = (const unsigned int*)(mask + (size_t)b * Tdim);
    int s = 0;
    for (int i = 0; i < Tdim / 4; ++i) s += __popc(m[i]);   // bool bytes are 0/1
    lenw[b] = (s < 1) ? 1 : s;
}

__global__ void prep_seg(const int* __restrict__ lenw, unsigned char* __restrict__ segw) {
    int idx = blockIdx.x * blockDim.x + threadIdx.x;
    if (idx >= Bdim * Tdim) return;
    int b = idx / Tdim, t = idx % Tdim;
    int s = (t * Sdim) / lenw[b];
    segw[idx] = (unsigned char)(s > (Sdim - 1) ? (Sdim - 1) : s);
}

__global__ void prep_w(const float* __restrict__ W_ih, const float* __restrict__ W_hh,
                       const float* __restrict__ b_ih, const float* __restrict__ b_hh,
                       ushort_t* __restrict__ Wc, float* __restrict__ biasw) {
    int idx = blockIdx.x * blockDim.x + threadIdx.x;
    const int total = Sdim * GATES * Kdim;
    if (idx < total) {
        int k = idx % Kdim;
        int r = (idx / Kdim) % GATES;
        int s = idx / (Kdim * GATES);
        float v = (k < Idim) ? W_ih[((size_t)s * GATES + r) * Idim + k]
                             : W_hh[((size_t)s * GATES + r) * Hdim + (k - Idim)];
        Wc[idx] = f2bf(v);
    }
    if (idx < Sdim * GATES) biasw[idx] = b_ih[idx] + b_hh[idx];
}

__global__ void prep_x(const float* __restrict__ x, ushort_t* __restrict__ xbf, int total) {
    int idx = blockIdx.x * blockDim.x + threadIdx.x;
    int stride = gridDim.x * blockDim.x;
    for (int i = idx; i < total; i += stride) xbf[i] = f2bf(x[i]);
}

// ---------------------------------------------------------------------------
// Main persistent LSTM kernel: 16 blocks x 1024 threads (32 waves / WGP)
// ---------------------------------------------------------------------------
__global__ __launch_bounds__(1024, 1)
void shift_lstm_main(const ushort_t* __restrict__ Wc,
                     const float* __restrict__ biasw,
                     const int* __restrict__ lenw,
                     const unsigned char* __restrict__ segw,
                     const ushort_t* __restrict__ xbf,
                     const float* __restrict__ xf32,
                     float* __restrict__ out,
                     float* __restrict__ lastout,
                     int use_tdm)
{
    extern __shared__ char smem[];
    float*         G     = (float*)(smem + G_OFF);       // [NT][GSTR] gates f32
    ushort_t*      hb    = (ushort_t*)(smem + H_OFF);    // [NT][HSTR] h bf16 (k-major rows)
    ushort_t*      xb0   = (ushort_t*)(smem + X0_OFF);   // [NT][XSTR] x bf16 buf0
    ushort_t*      xb1   = (ushort_t*)(smem + X1_OFF);   // buf1
    float*         bsh   = (float*)(smem + BIAS_OFF);    // [S][GATES]
    int*           lensh = (int*)(smem + LEN_OFF);       // [NT]
    unsigned char* segsh = (unsigned char*)(smem + SEG_OFF); // [2][NT]

    const int tid  = threadIdx.x;
    const int b0   = blockIdx.x * NT;
    const int lane = tid & 31;
    const int wid  = tid >> 5;
    const int lm   = lane & 15;   // A: M row / B,C,D: N column (batch within tile)
    const int lh   = lane >> 4;   // half selector

    // -------- setup --------
    for (int i = tid; i < NT * HSTR; i += 1024) hb[i] = 0;          // h0 = 0
    for (int i = tid; i < Sdim * GATES; i += 1024) bsh[i] = biasw[i];
    if (tid < NT) {
        int L = lenw[b0 + tid]; if (L < 1) L = 1;
        lensh[tid] = L;
        segsh[tid] = segw[(size_t)(b0 + tid) * Tdim + 0];
    }
    if (use_tdm) {
        if (wid == 0) {
            tdm_load_x(xbf + (size_t)(b0 * Tdim + 0) * Idim, X0_OFF);
            wait_tensor0();
        }
    } else {
        // fallback: threads convert x(t=0) f32 -> bf16 into xb0
        int e = tid * 2;
        int n2 = e >> 7, i2 = e & 127;
        const float* xp = xf32 + ((size_t)(b0 + n2) * Tdim + 0) * Idim + i2;
        xb0[n2 * XSTR + i2]     = f2bf(xp[0]);
        xb0[n2 * XSTR + i2 + 1] = f2bf(xp[1]);
    }
    __syncthreads();

    // per-thread cell state for update phase: batch un, 4 h-elements at uh
    const int un = tid >> 6;
    const int uh = (tid & 63) * 4;
    const int Lmy = lensh[un];
    float c0 = 0.f, c1 = 0.f, c2 = 0.f, c3 = 0.f;

    for (int t = 0; t < Tdim; ++t) {
        const int cur = t & 1;
        ushort_t* xbC = cur ? xb1 : xb0;
        ushort_t* xbN = cur ? xb0 : xb1;

        // prefetch x(t+1) into the other buffer via TDM (overlaps with compute)
        if (use_tdm && wid == 0 && (t + 1) < Tdim) {
            tdm_load_x(xbf + (size_t)(b0 * Tdim + (t + 1)) * Idim,
                       cur ? X0_OFF : X1_OFF);
        }

        // which segments are present in this batch tile at step t (uniform)
        const unsigned char* segc = segsh + cur * NT;
        int segmask = 0;
        #pragma unroll
        for (int n = 0; n < NT; ++n) segmask |= 1 << segc[n];

        // -------- GEMM: gates[n][m] = sum_k Wc[s][m][k] * hx[n][k] --------
        for (int s = 0; s < Sdim; ++s) {
            if (!((segmask >> s) & 1)) continue;
            #pragma unroll
            for (int mi = 0; mi < 2; ++mi) {
                const int mbase = (wid + mi * 32) * 16;
                v8f acc = {0.f, 0.f, 0.f, 0.f, 0.f, 0.f, 0.f, 0.f};
                const ushort_t* wrow =
                    Wc + (size_t)((s << 10) + mbase + lm) * Kdim + 8 * lh;
                #pragma unroll
                for (int kt = 0; kt < 12; ++kt) {
                    // A: 16x32 bf16, lane lm = row, chunks at k = 8*lh and 8*lh+16
                    v16bf Am = ld16(wrow + kt * 32, wrow + kt * 32 + 16);
                    // B: 32x16 bf16, lane lm = col, chunks at k = 8*lh and 8*lh+16
                    const ushort_t* bp = (kt < 4)
                        ? (xbC + lm * XSTR + kt * 32 + 8 * lh)
                        : (hb  + lm * HSTR + (kt - 4) * 32 + 8 * lh);
                    v16bf Bm = ld16(bp, bp + 16);
                    acc = __builtin_amdgcn_wmma_f32_16x16x32_bf16(
                              false, Am, false, Bm, (short)0, acc, false, false);
                }
                // D: lane lm = column n, VGPR r -> m = mbase + 8*lh + r
                if (segc[lm] == (unsigned char)s) {
                    float* gp = G + lm * GSTR + mbase + 8 * lh;
                    *(float4*)gp       = *((float4*)&acc);
                    *((float4*)gp + 1) = *(((float4*)&acc) + 1);
                }
            }
        }
        __syncthreads();

        // -------- LSTM cell update --------
        {
            const unsigned char* segc2 = segsh + cur * NT;
            const int sg = segc2[un];
            const float* bb = bsh + sg * GATES;
            const float* Gr = G + un * GSTR;
            float4 gi = *(const float4*)(Gr + uh);
            float4 gf = *(const float4*)(Gr + 256 + uh);
            float4 gg = *(const float4*)(Gr + 512 + uh);
            float4 go = *(const float4*)(Gr + 768 + uh);
            float4 bi = *(const float4*)(bb + uh);
            float4 bf = *(const float4*)(bb + 256 + uh);
            float4 bg = *(const float4*)(bb + 512 + uh);
            float4 bo = *(const float4*)(bb + 768 + uh);
            gi.x += bi.x; gi.y += bi.y; gi.z += bi.z; gi.w += bi.w;
            gf.x += bf.x; gf.y += bf.y; gf.z += bf.z; gf.w += bf.w;
            gg.x += bg.x; gg.y += bg.y; gg.z += bg.z; gg.w += bg.w;
            go.x += bo.x; go.y += bo.y; go.z += bo.z; go.w += bo.w;

            float cn0 = sigm(gf.x) * c0 + sigm(gi.x) * tanh_f(gg.x);
            float cn1 = sigm(gf.y) * c1 + sigm(gi.y) * tanh_f(gg.y);
            float cn2 = sigm(gf.z) * c2 + sigm(gi.z) * tanh_f(gg.z);
            float cn3 = sigm(gf.w) * c3 + sigm(gi.w) * tanh_f(gg.w);
            float hn0 = sigm(go.x) * tanh_f(cn0);
            float hn1 = sigm(go.y) * tanh_f(cn1);
            float hn2 = sigm(go.z) * tanh_f(cn2);
            float hn3 = sigm(go.w) * tanh_f(cn3);

            const bool valid = (t < Lmy);
            if (valid) {
                c0 = cn0; c1 = cn1; c2 = cn2; c3 = cn3;
                ushort4 hv;
                hv.x = f2bf(hn0); hv.y = f2bf(hn1);
                hv.z = f2bf(hn2); hv.w = f2bf(hn3);
                *(ushort4*)(hb + un * HSTR + uh) = hv;   // h for next step's GEMM
            }
            float4 ov;
            ov.x = valid ? hn0 : 0.f; ov.y = valid ? hn1 : 0.f;
            ov.z = valid ? hn2 : 0.f; ov.w = valid ? hn3 : 0.f;
            *(float4*)(out + ((size_t)(b0 + un) * Tdim + t) * Hdim + uh) = ov;
            if (t == Lmy - 1) {
                *(float4*)(lastout + (size_t)(b0 + un) * Hdim + uh) =
                    make_float4(hn0, hn1, hn2, hn3);
            }
        }
        // prefetch seg(t+1)
        if (tid < NT && (t + 1) < Tdim)
            segsh[(1 - cur) * NT + tid] = segw[(size_t)(b0 + tid) * Tdim + t + 1];
        // fallback x(t+1) prefetch
        if (!use_tdm && (t + 1) < Tdim) {
            int e = tid * 2;
            int n2 = e >> 7, i2 = e & 127;
            const float* xp = xf32 + ((size_t)(b0 + n2) * Tdim + (t + 1)) * Idim + i2;
            xbN[n2 * XSTR + i2]     = f2bf(xp[0]);
            xbN[n2 * XSTR + i2 + 1] = f2bf(xp[1]);
        }
        if (use_tdm && wid == 0 && (t + 1) < Tdim) wait_tensor0();
        __syncthreads();
    }
}

// ---------------------------------------------------------------------------
extern "C" void kernel_launch(void* const* d_in, const int* in_sizes, int n_in,
                              void* d_out, int out_size, void* d_ws, size_t ws_size,
                              hipStream_t stream) {
    (void)in_sizes; (void)n_in; (void)out_size;
    const float*         x    = (const float*)d_in[0];
    const unsigned char* mask = (const unsigned char*)d_in[1];   // jnp bool -> 1 byte
    const float*         W_ih = (const float*)d_in[2];
    const float*         W_hh = (const float*)d_in[3];
    const float*         b_ih = (const float*)d_in[4];
    const float*         b_hh = (const float*)d_in[5];
    float* out  = (float*)d_out;
    float* last = out + (size_t)Bdim * Tdim * Hdim;

    char* ws = (char*)d_ws;
    size_t off = 0;
    ushort_t* Wc = (ushort_t*)(ws + off); off += (size_t)Sdim * GATES * Kdim * 2;
    float* biasw = (float*)(ws + off);    off += (size_t)Sdim * GATES * 4;
    int* lenw = (int*)(ws + off);         off += (size_t)Bdim * 4;
    unsigned char* segw = (unsigned char*)(ws + off); off += (size_t)Bdim * Tdim;
    off = (off + 255) & ~(size_t)255;
    ushort_t* xbf = (ushort_t*)(ws + off);
    size_t need_tdm = off + (size_t)Bdim * Tdim * Idim * 2;
    int use_tdm = (ws_size >= need_tdm) ? 1 : 0;

    prep_len<<<1, 256, 0, stream>>>(mask, lenw);
    prep_seg<<<(Bdim * Tdim + 255) / 256, 256, 0, stream>>>(lenw, segw);
    prep_w<<<(Sdim * GATES * Kdim + 255) / 256, 256, 0, stream>>>(W_ih, W_hh, b_ih, b_hh, Wc, biasw);
    if (use_tdm)
        prep_x<<<4096, 1024, 0, stream>>>(x, xbf, Bdim * Tdim * Idim);

    (void)hipFuncSetAttribute((const void*)shift_lstm_main,
                              hipFuncAttributeMaxDynamicSharedMemorySize, SMEM_BYTES);
    shift_lstm_main<<<Bdim / NT, 1024, SMEM_BYTES, stream>>>(
        Wc, biasw, lenw, segw, xbf, x, out, last, use_tdm);
}
